// GPTLanguageModel_69063074119757
// MI455X (gfx1250) — matrix operand order repaired and checked by reading, compiled, not run on hardware
//
#include <hip/hip_runtime.h>
#include <hip/hip_bf16.h>
#include <stdint.h>

typedef __attribute__((ext_vector_type(16))) _Float16 v16h;
typedef __attribute__((ext_vector_type(8)))  float    v8f;
typedef __attribute__((ext_vector_type(4)))  unsigned int u32x4;
typedef __attribute__((ext_vector_type(8)))  int      i32x8;
typedef __attribute__((ext_vector_type(4)))  int      i32x4;

#define BB   128
#define TT   256
#define MM   32768      // B*T
#define CC   384
#define HH   6
#define HSZ  64
#define FFD  1536
#define NL   6
#define VV   65

#if defined(__HIP_DEVICE_COMPILE__) && __has_builtin(__builtin_amdgcn_tensor_load_to_lds)
#define HAVE_TDM 1
#else
#define HAVE_TDM 0
#endif

// ---------------------------------------------------------------------------
// Tensor Data Mover: 2-D f32 tile Global -> LDS (ISA cdna5 ch.8, D# groups).
//   lds_off            : byte offset of tile start in LDS
//   gsrc               : global address of tile start
//   tile_w/tile_h      : tile dims (elements / rows)
//   stride_elems       : tensor_dim0_stride (row pitch in elements)
//   tensor_w/tensor_h  : remaining valid tensor extent from tile start
//                        (reads beyond return zero -> hardware ragged-N fill)
//   pad_iv/pad_am      : LDS pad codes: pad 2^(iv+1) dwords -> insert
//                        (am+1) dwords (gives our bank-conflict padding)
// ---------------------------------------------------------------------------
#if HAVE_TDM
__device__ __forceinline__ void tdm_load_tile_f32(
    unsigned lds_off, const float* gsrc,
    unsigned tile_w, unsigned tile_h,
    unsigned long stride_elems,
    unsigned long tensor_w, unsigned long tensor_h,
    unsigned pad_iv, unsigned pad_am)
{
    const unsigned long ga = (unsigned long)(uintptr_t)gsrc;
    u32x4 g0;
    g0.x = 1u;                                            // count=1, user mode
    g0.y = lds_off;                                       // lds_addr [63:32]
    g0.z = (unsigned)(ga & 0xffffffffu);                  // global_addr[31:0]
    g0.w = (unsigned)((ga >> 32) & 0x01ffffffu)           // global_addr[56:32]
         | 0x80000000u;                                   // type=2 ("image")
    i32x8 g1;
    g1[0] = (int)((2u << 16)                              // data_size = 4B
                | (1u << 20)                              // pad_enable
                | (pad_iv << 22) | (pad_am << 25));
    g1[1] = (int)((tensor_w & 0xffffu) << 16);            // tensor_dim0 lo16
    g1[2] = (int)(((tensor_w >> 16) & 0xffffu)            // tensor_dim0 hi16
                | ((tensor_h & 0xffffu) << 16));          // tensor_dim1 lo16
    g1[3] = (int)(((tensor_h >> 16) & 0xffffu)            // tensor_dim1 hi16
                | ((tile_w & 0xffffu) << 16));            // tile_dim0
    g1[4] = (int)(tile_h & 0xffffu);                      // tile_dim1 (dim2=0)
    g1[5] = (int)(unsigned)(stride_elems & 0xffffffffu);  // dim0_stride lo32
    g1[6] = (int)((unsigned)((stride_elems >> 32) & 0xffffu)); // hi16
    g1[7] = 0;
    const i32x4 gz = {0, 0, 0, 0};
#if defined(__clang_major__) && __clang_major__ >= 23
    const i32x8 gz8 = {0, 0, 0, 0, 0, 0, 0, 0};
    __builtin_amdgcn_tensor_load_to_lds(g0, g1, gz, gz, gz8, 0);
#else
    __builtin_amdgcn_tensor_load_to_lds(g0, g1, gz, gz, 0);
#endif
}
#endif

// ---------------------------------------------------------------------------
// Embedding gather: x[m, :] = emb[idx[m], :]
// ---------------------------------------------------------------------------
__global__ void __launch_bounds__(128) embed_kernel(const int* __restrict__ idx,
                                                    const float* __restrict__ emb,
                                                    float* __restrict__ x) {
    const int row = blockIdx.x;
    const int tok = idx[row];
    for (int i = threadIdx.x; i < CC; i += 128)
        x[(long)row * CC + i] = emb[(long)tok * CC + i];
}

// ---------------------------------------------------------------------------
// LayerNorm: one block (128 threads) per token, C = 384 -> 3 elems/thread
// ---------------------------------------------------------------------------
__global__ void __launch_bounds__(128) ln_kernel(const float* __restrict__ x,
                                                 const float* __restrict__ g,
                                                 const float* __restrict__ b,
                                                 float* __restrict__ out) {
    __shared__ float s1[128], s2[128];
    const int row = blockIdx.x;
    const int t = threadIdx.x;
    const float* xr = x + (long)row * CC;
    float v0 = xr[t], v1 = xr[t + 128], v2 = xr[t + 256];
    s1[t] = v0 + v1 + v2;
    s2[t] = v0 * v0 + v1 * v1 + v2 * v2;
    __syncthreads();
    for (int off = 64; off > 0; off >>= 1) {
        if (t < off) { s1[t] += s1[t + off]; s2[t] += s2[t + off]; }
        __syncthreads();
    }
    const float mean = s1[0] * (1.0f / CC);
    const float var  = s2[0] * (1.0f / CC) - mean * mean;
    const float rs   = rsqrtf(var + 1e-5f);
    float* orow = out + (long)row * CC;
    orow[t]       = (v0 - mean) * rs * g[t]       + b[t];
    orow[t + 128] = (v1 - mean) * rs * g[t + 128] + b[t + 128];
    orow[t + 256] = (v2 - mean) * rs * g[t + 256] + b[t + 256];
}

// ---------------------------------------------------------------------------
// WMMA GEMM with TDM double-buffered LDS staging:
//   out = A[M,K] * W[K,N] + bias (+resid) (relu)
// Block: 128 threads (4 waves), tile 64x64, K-step 32.
// Wave 0 issues tensor_load_to_lds for the next K-step while all 4 waves run
// v_wmma_f32_16x16x32_f16 on the current tiles (TENSORcnt-pipelined).
// TDM pad_enable reproduces the padded [64][34] / [32][66] LDS layouts.
// ---------------------------------------------------------------------------
template <bool RELU, bool RES>
__global__ void __launch_bounds__(128) gemm_kernel(
    const float* __restrict__ A, const float* __restrict__ W,
    const float* __restrict__ bias, const float* __restrict__ resid,
    float* __restrict__ out, int M, int N, int K,
    long wZ, long bZ, long oZ)
{
    __shared__ float Af[2][64][34];   // [m][k], pad 32->34 dwords
    __shared__ float Wf[2][32][66];   // [k][n], pad 64->66 dwords

    const int tid  = threadIdx.x;
    const int wave = tid >> 5;
    const int lane = tid & 31;
    const int hi   = lane >> 4;
    const int ln   = lane & 15;
    const int z    = blockIdx.z;

    const float* Wz = W    + (long)z * wZ;
    const float* bz = bias + (long)z * bZ;
    float*       oz = out  + (long)z * oZ;

    const int m0 = blockIdx.y * 64;
    const int n0 = blockIdx.x * 64;
    const int ksteps = K / 32;

    v8f acc[4] = {};

#if HAVE_TDM
    if (tid < 32) {   // wave 0: prologue DMA of K-step 0 (A then W)
        tdm_load_tile_f32((unsigned)(uintptr_t)&Af[0][0][0],
                          A + (long)m0 * K, 32, 64,
                          (unsigned long)K, (unsigned long)K,
                          (unsigned long)(M - m0), 4u, 1u);
        tdm_load_tile_f32((unsigned)(uintptr_t)&Wf[0][0][0],
                          Wz + (long)n0, 64, 32,
                          (unsigned long)N, (unsigned long)(N - n0),
                          (unsigned long)K, 5u, 1u);
    }
#else
    for (int i = tid; i < 64 * 32; i += 128) {
        const int r = i >> 5, c = i & 31;
        Af[0][r][c] = A[(long)(m0 + r) * K + c];
    }
    for (int i = tid; i < 32 * 64; i += 128) {
        const int r = i >> 6, c = i & 63;
        const int col = n0 + c;
        Wf[0][r][c] = (col < N) ? Wz[(long)r * N + col] : 0.0f;
    }
#endif

    for (int s = 0; s < ksteps; ++s) {
        const int cur = s & 1;
#if HAVE_TDM
        if (tid < 32) __builtin_amdgcn_s_wait_tensorcnt(0);
#endif
        __syncthreads();

        if (s + 1 < ksteps) {
            const int k1 = (s + 1) * 32;
#if HAVE_TDM
            if (tid < 32) {   // prefetch next K-step into the other buffer
                tdm_load_tile_f32((unsigned)(uintptr_t)&Af[cur ^ 1][0][0],
                                  A + (long)m0 * K + k1, 32, 64,
                                  (unsigned long)K, (unsigned long)(K - k1),
                                  (unsigned long)(M - m0), 4u, 1u);
                tdm_load_tile_f32((unsigned)(uintptr_t)&Wf[cur ^ 1][0][0],
                                  Wz + (long)k1 * N + n0, 64, 32,
                                  (unsigned long)N, (unsigned long)(N - n0),
                                  (unsigned long)(K - k1), 5u, 1u);
            }
#else
            for (int i = tid; i < 64 * 32; i += 128) {
                const int r = i >> 5, c = i & 31;
                Af[cur ^ 1][r][c] = A[(long)(m0 + r) * K + k1 + c];
            }
            for (int i = tid; i < 32 * 64; i += 128) {
                const int r = i >> 6, c = i & 63;
                const int col = n0 + c;
                Wf[cur ^ 1][r][c] = (col < N) ? Wz[(long)(k1 + r) * N + col] : 0.0f;
            }
#endif
        }

        // A fragment: 16x32 f16, ISA A-layout (pairs contiguous in K)
        v16h af;
#pragma unroll
        for (int i = 0; i < 16; ++i) {
            const int vg = i >> 1, sub = i & 1;
            const int kk = ((vg < 4) ? 0 : 16) + hi * 8 + (vg & 3) * 2 + sub;
            af[i] = (_Float16)Af[cur][wave * 16 + ln][kk];
        }
#pragma unroll
        for (int nt = 0; nt < 4; ++nt) {
            v16h bf;   // B fragment: 32x16, K = hi*8 + slot (+8 if slot>=8), N = ln
#pragma unroll
            for (int i = 0; i < 16; ++i)
                bf[i] = (_Float16)Wf[cur][hi * 8 + i + ((i >= 8) ? 8 : 0)][nt * 16 + ln];
            acc[nt] = __builtin_amdgcn_wmma_f32_16x16x32_f16(
                false, af, false, bf, (short)0, acc[nt], false, false);
        }
        __syncthreads();
    }

    // C layout: row = hi*8 + r, col = ln
#pragma unroll
    for (int nt = 0; nt < 4; ++nt) {
        const int col = n0 + nt * 16 + ln;
        if (col >= N) continue;
        const float bv = bz[col];
#pragma unroll
        for (int r = 0; r < 8; ++r) {
            const int row = m0 + wave * 16 + hi * 8 + r;
            if (row >= M) continue;
            float val = acc[nt][r] + bv;
            if (RES) val += resid[(long)row * N + col];
            if (RELU) val = fmaxf(val, 0.0f);
            oz[(long)row * N + col] = val;
        }
    }
}

// ---------------------------------------------------------------------------
// Flash attention, causal. 1 wave per (qt, h, b); 16 query rows per block.
// q/k/v layout: [H][M][64]. Output o: [M, 384] (col offset h*64).
// Key blocks of 32: S = Q*K^T (2 WMMA per 16x16 tile), online softmax with
// __shfl_xor row reductions, P restaged through LDS (C-layout -> A-layout),
// then O += P*V (4 WMMA).
// ---------------------------------------------------------------------------
__global__ void __launch_bounds__(32) attn_kernel(
    const float* __restrict__ q, const float* __restrict__ k,
    const float* __restrict__ v, float* __restrict__ o)
{
    __shared__ _Float16 Ks[32][66];   // [key][channel]
    __shared__ _Float16 Vs[32][66];   // [key][channel]
    __shared__ _Float16 Ps[16][34];   // [qrow][key_local]

    const int lane = threadIdx.x;
    const int hi = lane >> 4, ln = lane & 15;
    const int qt = blockIdx.x;        // 0..15
    const int h  = blockIdx.y;
    const int b  = blockIdx.z;

    const float* qp = q + (long)h * MM * HSZ;
    const float* kp = k + (long)h * MM * HSZ;
    const float* vp = v + (long)h * MM * HSZ;
    const int m0 = b * TT + qt * 16;  // global row base

    v16h qf0, qf1;
#pragma unroll
    for (int i = 0; i < 16; ++i) {
        const int vg = i >> 1, sub = i & 1;
        const int kk = ((vg < 4) ? 0 : 16) + hi * 8 + (vg & 3) * 2 + sub;
        qf0[i] = (_Float16)qp[(long)(m0 + ln) * HSZ + kk];
        qf1[i] = (_Float16)qp[(long)(m0 + ln) * HSZ + 32 + kk];
    }

    float mrow[8], lrow[8];
    v8f accO[4] = {};
#pragma unroll
    for (int r = 0; r < 8; ++r) { mrow[r] = -1e30f; lrow[r] = 0.0f; }

    const int nkb = (qt * 16 + 15) / 32 + 1;
    for (int kb = 0; kb < nkb; ++kb) {
        const int kbase = kb * 32;
        __syncthreads();
        for (int i = lane; i < 32 * 64; i += 32) {
            const int r = i >> 6, c = i & 63;
            const int key = kbase + r;
            _Float16 kx = (_Float16)0.0f, vx = (_Float16)0.0f;
            if (key < TT) {
                kx = (_Float16)kp[(long)(b * TT + key) * HSZ + c];
                vx = (_Float16)vp[(long)(b * TT + key) * HSZ + c];
            }
            Ks[r][c] = kx; Vs[r][c] = vx;
        }
        __syncthreads();

        float sv[2][8];
#pragma unroll
        for (int nt = 0; nt < 2; ++nt) {
            v16h bf0, bf1;   // B = K^T: [channel][key]
#pragma unroll
            for (int i = 0; i < 16; ++i) {
                bf0[i] = Ks[nt * 16 + ln][hi * 8 + i + ((i >= 8) ? 8 : 0)];
                bf1[i] = Ks[nt * 16 + ln][32 + hi * 8 + i + ((i >= 8) ? 8 : 0)];
            }
            v8f s = {};
            s = __builtin_amdgcn_wmma_f32_16x16x32_f16(false, qf0, false, bf0,
                                                       (short)0, s, false, false);
            s = __builtin_amdgcn_wmma_f32_16x16x32_f16(false, qf1, false, bf1,
                                                       (short)0, s, false, false);
#pragma unroll
            for (int r = 0; r < 8; ++r) {
                const int row_t = qt * 16 + hi * 8 + r;
                const int key_t = kbase + nt * 16 + ln;
                const float sx = s[r] * 0.125f;     // HS^-0.5
                sv[nt][r] = (key_t <= row_t) ? sx : -1e30f;
            }
        }

        // online softmax per row (rows live in 16-lane halves)
#pragma unroll
        for (int r = 0; r < 8; ++r) {
            float tm = fmaxf(sv[0][r], sv[1][r]);
            for (int off = 1; off < 16; off <<= 1)
                tm = fmaxf(tm, __shfl_xor(tm, off, 16));
            const float mnew = fmaxf(mrow[r], tm);
            const float corr = __expf(mrow[r] - mnew);
            const float p0 = __expf(sv[0][r] - mnew);
            const float p1 = __expf(sv[1][r] - mnew);
            float rs = p0 + p1;
            for (int off = 1; off < 16; off <<= 1)
                rs += __shfl_xor(rs, off, 16);
            lrow[r] = lrow[r] * corr + rs;
            mrow[r] = mnew;
#pragma unroll
            for (int nt = 0; nt < 4; ++nt) accO[nt][r] *= corr;
            Ps[hi * 8 + r][ln]      = (_Float16)p0;
            Ps[hi * 8 + r][16 + ln] = (_Float16)p1;
        }
        __syncthreads();

        v16h pf;   // P (16x32) as A-frag from LDS
#pragma unroll
        for (int i = 0; i < 16; ++i) {
            const int vg = i >> 1, sub = i & 1;
            const int kk = ((vg < 4) ? 0 : 16) + hi * 8 + (vg & 3) * 2 + sub;
            pf[i] = Ps[ln][kk];
        }
#pragma unroll
        for (int nt = 0; nt < 4; ++nt) {
            v16h vf;   // B = V: [key][channel tile nt]
#pragma unroll
            for (int i = 0; i < 16; ++i) vf[i] = Vs[hi * 8 + i + ((i >= 8) ? 8 : 0)][nt * 16 + ln];
            accO[nt] = __builtin_amdgcn_wmma_f32_16x16x32_f16(
                false, pf, false, vf, (short)0, accO[nt], false, false);
        }
    }

#pragma unroll
    for (int nt = 0; nt < 4; ++nt) {
        const int col = h * HSZ + nt * 16 + ln;
#pragma unroll
        for (int r = 0; r < 8; ++r) {
            const int row = m0 + hi * 8 + r;
            o[(long)row * CC + col] = accO[nt][r] / lrow[r];
        }
    }
}

// ---------------------------------------------------------------------------
// Host launcher
// ---------------------------------------------------------------------------
extern "C" void kernel_launch(void* const* d_in, const int* in_sizes, int n_in,
                              void* d_out, int out_size, void* d_ws, size_t ws_size,
                              hipStream_t stream) {
    (void)in_sizes; (void)n_in; (void)out_size; (void)ws_size;
    const int*   idx   = (const int*)  d_in[0];
    const float* emb   = (const float*)d_in[1];
    const float* Wq    = (const float*)d_in[2];
    const float* bq    = (const float*)d_in[3];
    const float* Wk    = (const float*)d_in[4];
    const float* bk    = (const float*)d_in[5];
    const float* Wv    = (const float*)d_in[6];
    const float* bv    = (const float*)d_in[7];
    const float* Wo    = (const float*)d_in[8];
    const float* bo    = (const float*)d_in[9];
    const float* ln1g  = (const float*)d_in[10];
    const float* ln1b  = (const float*)d_in[11];
    const float* W1    = (const float*)d_in[12];
    const float* b1    = (const float*)d_in[13];
    const float* W2    = (const float*)d_in[14];
    const float* b2    = (const float*)d_in[15];
    const float* ln2g  = (const float*)d_in[16];
    const float* ln2b  = (const float*)d_in[17];
    const float* lnfg  = (const float*)d_in[18];
    const float* lnfb  = (const float*)d_in[19];
    const float* Whead = (const float*)d_in[20];
    const float* bhead = (const float*)d_in[21];
    float* logits = (float*)d_out;

    // workspace arena (~302 MB):
    //   x (50.3MB) | h (50.3MB) | arena: q|k|v|o (201.3MB) reused as mid
    const long XB = (long)MM * CC * sizeof(float);          // 50,331,648
    const long QB = (long)HH * MM * HSZ * sizeof(float);    // 50,331,648
    char* ws = (char*)d_ws;
    float* x    = (float*)ws;
    float* hbuf = (float*)(ws + XB);
    char*  ar   = ws + 2 * XB;
    float* qb   = (float*)(ar);
    float* kb   = (float*)(ar + QB);
    float* vb   = (float*)(ar + 2 * QB);
    float* ob   = (float*)(ar + 3 * QB);
    float* mid  = (float*)(ar);     // reuses q/k/v/o after attention+proj

    const dim3 blk128(128), blk32(32);

    embed_kernel<<<MM, blk128, 0, stream>>>(idx, emb, x);

    for (int l = 0; l < NL; ++l) {
        const float* Wq_l = Wq + (long)l * HH * CC * HSZ;
        const float* bq_l = bq + (long)l * HH * HSZ;
        const float* Wk_l = Wk + (long)l * HH * CC * HSZ;
        const float* bk_l = bk + (long)l * HH * HSZ;
        const float* Wv_l = Wv + (long)l * HH * CC * HSZ;
        const float* bv_l = bv + (long)l * HH * HSZ;
        const float* Wo_l = Wo + (long)l * CC * CC;
        const float* bo_l = bo + (long)l * CC;
        const float* W1_l = W1 + (long)l * CC * FFD;
        const float* b1_l = b1 + (long)l * FFD;
        const float* W2_l = W2 + (long)l * FFD * CC;
        const float* b2_l = b2 + (long)l * CC;

        // h = LN1(x)
        ln_kernel<<<MM, blk128, 0, stream>>>(x, ln1g + (long)l * CC,
                                             ln1b + (long)l * CC, hbuf);
        // q/k/v = h @ W{q,k,v}[h] + b   (grid.z = heads)
        gemm_kernel<false, false><<<dim3(1, MM / 64, HH), blk128, 0, stream>>>(
            hbuf, Wq_l, bq_l, nullptr, qb, MM, HSZ, CC,
            (long)CC * HSZ, (long)HSZ, (long)MM * HSZ);
        gemm_kernel<false, false><<<dim3(1, MM / 64, HH), blk128, 0, stream>>>(
            hbuf, Wk_l, bk_l, nullptr, kb, MM, HSZ, CC,
            (long)CC * HSZ, (long)HSZ, (long)MM * HSZ);
        gemm_kernel<false, false><<<dim3(1, MM / 64, HH), blk128, 0, stream>>>(
            hbuf, Wv_l, bv_l, nullptr, vb, MM, HSZ, CC,
            (long)CC * HSZ, (long)HSZ, (long)MM * HSZ);
        // o = causal softmax(q k^T / 8) v
        attn_kernel<<<dim3(TT / 16, HH, BB), blk32, 0, stream>>>(qb, kb, vb, ob);
        // x = x + o @ Wo + bo
        gemm_kernel<false, true><<<dim3(CC / 64, MM / 64, 1), blk128, 0, stream>>>(
            ob, Wo_l, bo_l, x, x, MM, CC, CC, 0, 0, 0);
        // h = LN2(x)
        ln_kernel<<<MM, blk128, 0, stream>>>(x, ln2g + (long)l * CC,
                                             ln2b + (long)l * CC, hbuf);
        // mid = relu(h @ W1 + b1)
        gemm_kernel<true, false><<<dim3(FFD / 64, MM / 64, 1), blk128, 0, stream>>>(
            hbuf, W1_l, b1_l, nullptr, mid, MM, FFD, CC, 0, 0, 0);
        // x = x + mid @ W2 + b2
        gemm_kernel<false, true><<<dim3(CC / 64, MM / 64, 1), blk128, 0, stream>>>(
            mid, W2_l, b2_l, x, x, MM, CC, FFD, 0, 0, 0);
    }

    // final LN + head (N = 65, ragged -> TDM zero-fills OOB columns)
    ln_kernel<<<MM, blk128, 0, stream>>>(x, lnfg, lnfb, hbuf);
    gemm_kernel<false, false><<<dim3((VV + 63) / 64, MM / 64, 1), blk128, 0, stream>>>(
        hbuf, Whead, bhead, nullptr, logits, MM, VV, CC, 0, 0, 0);
}
